// Attention_68135361184467
// MI455X (gfx1250) — matrix operand order repaired and checked by reading, not compiled, not run on hardware
//
#include <hip/hip_runtime.h>

// Attention: B=2,H=16,S=4096,D=64, fp32 in/out. d_out = out[B,H,S,D] ++ weights[B,H,S,S].
#define SDIM 4096
#define DDIM 64
#define BHN  32
#define QT   128          // query rows per block (8 waves x 16)
#define KTILE 32          // keys per iteration
#define KSTR 72           // LDS row stride (u16) for K tile: 64 + 8 pad (16B aligned, conflict-free)
#define VSTR 40           // LDS row stride (u16) for V^T tile: 32 + 8 pad
#define WSTR 40           // LDS row stride (u16) for per-wave weight staging

typedef __attribute__((ext_vector_type(16))) __bf16          v16bf;
typedef __attribute__((ext_vector_type(8)))  float           v8f;
typedef __attribute__((ext_vector_type(4)))  float           v4f;
typedef __attribute__((ext_vector_type(16))) unsigned short  v16us;
typedef __attribute__((ext_vector_type(8)))  unsigned short  v8us;

__device__ __forceinline__ unsigned short f2bf(float f) {
  unsigned u = __builtin_bit_cast(unsigned, f);
  u += 0x7FFFu + ((u >> 16) & 1u);          // round-to-nearest-even
  return (unsigned short)(u >> 16);
}

__device__ __forceinline__ v16bf lds_bf16x16(const unsigned short* p) {
  return __builtin_bit_cast(v16bf, *(const v16us*)p);
}

__global__ __launch_bounds__(256)
void attn_fused(const float* __restrict__ Qg, const float* __restrict__ Kg,
                const float* __restrict__ Vg, float* __restrict__ Og,
                float* __restrict__ Wg)
{
  __shared__ unsigned short Kt[KTILE * KSTR];     // K tile, bf16, row-major [key][d]
  __shared__ unsigned short Vt[DDIM * VSTR];      // V tile transposed, bf16 [d][key]
  __shared__ unsigned short Wst[8 * 16 * WSTR];   // per-wave weight staging [16][32]

  const int tid  = threadIdx.x;
  const int wave = tid >> 5;
  const int lane = tid & 31;
  const int half = lane >> 4;     // 0: lanes 0-15, 1: lanes 16-31
  const int ln   = lane & 15;

  const int q0b  = blockIdx.x * QT;
  const int bh   = blockIdx.y;
  const int kend = q0b + QT;                 // causal: keys >= kend are all masked for this block
  const int qw   = q0b + wave * 16;          // this wave's first query row
  // Scalar (SGPR) copy of the wave-uniform first query row: makes the per-tile
  // diagonal test a uniform s_cbranch instead of a divergent exec-mask dance.
  const int qws  = __builtin_amdgcn_readfirstlane(qw);

  const float* Qp = Qg + (size_t)bh * SDIM * DDIM;
  const float* Kp = Kg + (size_t)bh * SDIM * DDIM;
  const float* Vp = Vg + (size_t)bh * SDIM * DDIM;
  float*       Op = Og + (size_t)bh * SDIM * DDIM;
  float*       Wp = Wg + (size_t)bh * (size_t)SDIM * SDIM;

  // ---- Q A-tiles (16 rows x 32 d, two d-chunks), pre-scaled by 1/(Temp*sqrt(D)) ----
  // A operand, this code's k order (B uses the same): lane holds row (ln); elems 0..15 = K 16*half+j
  v16bf aq[2];
  {
    const float scale = 0.125f;   // 1/sqrt(64), Temp=1
    const float* qr = Qp + (size_t)(qw + ln) * DDIM;
    #pragma unroll
    for (int c = 0; c < 2; ++c) {
      v16us bits;
      const int base = 32 * c + 16 * half;
      #pragma unroll
      for (int j = 0; j < 8; ++j) {
        bits[j]     = f2bf(qr[base + j]      * scale);
        bits[8 + j] = f2bf(qr[base + 8 + j] * scale);
      }
      aq[c] = __builtin_bit_cast(v16bf, bits);
    }
  }

  // Per-elem row indices (C/D layout: row = qw + r + 8*half, col = ln)
  int rowi[8];
  #pragma unroll
  for (int r = 0; r < 8; ++r) rowi[r] = qw + r + 8 * half;

  // Per-lane partial sums of exp(score) (softmax is shift-invariant; scores are O(5),
  // so unshifted exp is safe in fp32 and masked entries underflow to exactly 0).
  float ssum[8];
  #pragma unroll
  for (int r = 0; r < 8; ++r) ssum[r] = 0.f;

  const int str = tid >> 3;          // staging row 0..31
  const int stc = (tid & 7) * 8;     // staging col base 0..56

  // ================= Pass A: per-row sum of exp(score), no cross-lane ops in loop =================
  for (int kb = 0; kb < kend; kb += KTILE) {
    __syncthreads();
    {
      const float* g = Kp + (size_t)(kb + str) * DDIM + stc;
      v8us h;
      #pragma unroll
      for (int j = 0; j < 8; ++j) h[j] = f2bf(g[j]);
      *(v8us*)&Kt[str * KSTR + stc] = h;
      if (kb + KTILE < kend && tid < 64)
        __builtin_prefetch(Kp + (size_t)(kb + KTILE) * DDIM + tid * 32, 0, 1);
    }
    __syncthreads();

    #pragma unroll
    for (int kh = 0; kh < 2; ++kh) {
      v8f acc = {};
      const int klocal = kh * 16 + ln;   // B layout: lane = col (key); elems = 16 contiguous d rows
      #pragma unroll
      for (int c = 0; c < 2; ++c) {
        v16bf b = lds_bf16x16(&Kt[klocal * KSTR + 32 * c + 16 * half]);
        acc = __builtin_amdgcn_wmma_f32_16x16x32_bf16(false, aq[c], false, b,
                                                      (short)0, acc, false, false);
      }
      if (kb + KTILE - 1 > qws) {        // uniform scalar branch (tile touches causal boundary)
        const int kcol = kb + kh * 16 + ln;
        #pragma unroll
        for (int r = 0; r < 8; ++r)
          ssum[r] += __expf(acc[r] + ((kcol > rowi[r]) ? -1.0e9f : 0.f));
      } else {
        #pragma unroll
        for (int r = 0; r < 8; ++r)
          ssum[r] += __expf(acc[r]);
      }
    }
  }

  // One-shot reduction across the 16 lanes sharing each row, then invert.
  float inv[8];
  #pragma unroll
  for (int r = 0; r < 8; ++r) {
    float s = ssum[r];
    s += __shfl_xor(s, 1);
    s += __shfl_xor(s, 2);
    s += __shfl_xor(s, 4);
    s += __shfl_xor(s, 8);
    inv[r] = 1.0f / s;
  }

  // ================= Pass B: write weights (NT stores) + accumulate PV =================
  v8f acco[4] = {};
  unsigned short* ws = &Wst[wave * 16 * WSTR];

  for (int kb = 0; kb < kend; kb += KTILE) {
    __syncthreads();
    {
      const float* g = Kp + (size_t)(kb + str) * DDIM + stc;
      v8us h;
      #pragma unroll
      for (int j = 0; j < 8; ++j) h[j] = f2bf(g[j]);
      *(v8us*)&Kt[str * KSTR + stc] = h;
      const float* gv = Vp + (size_t)(kb + str) * DDIM + stc;
      #pragma unroll
      for (int j = 0; j < 8; ++j) Vt[(stc + j) * VSTR + str] = f2bf(gv[j]);
      if (kb + KTILE < kend && tid < 64) {
        __builtin_prefetch(Kp + (size_t)(kb + KTILE) * DDIM + tid * 32, 0, 1);
        __builtin_prefetch(Vp + (size_t)(kb + KTILE) * DDIM + tid * 32, 0, 1);
      }
    }
    __syncthreads();

    #pragma unroll
    for (int kh = 0; kh < 2; ++kh) {
      v8f acc = {};
      const int klocal = kh * 16 + ln;
      #pragma unroll
      for (int c = 0; c < 2; ++c) {
        v16bf b = lds_bf16x16(&Kt[klocal * KSTR + 32 * c + 16 * half]);
        acc = __builtin_amdgcn_wmma_f32_16x16x32_bf16(false, aq[c], false, b,
                                                      (short)0, acc, false, false);
      }
      const int kcol = kb + kh * 16 + ln;
      if (kb + KTILE - 1 > qws) {        // uniform scalar branch
        #pragma unroll
        for (int r = 0; r < 8; ++r) {
          float sc  = acc[r] + ((kcol > rowi[r]) ? -1.0e9f : 0.f);
          float wgt = __expf(sc) * inv[r];
          __builtin_nontemporal_store(wgt, &Wp[(size_t)rowi[r] * SDIM + kcol]);
          ws[(r + 8 * half) * WSTR + kh * 16 + ln] = f2bf(wgt);
        }
      } else {
        #pragma unroll
        for (int r = 0; r < 8; ++r) {
          float wgt = __expf(acc[r]) * inv[r];
          __builtin_nontemporal_store(wgt, &Wp[(size_t)rowi[r] * SDIM + kcol]);
          ws[(r + 8 * half) * WSTR + kh * 16 + ln] = f2bf(wgt);
        }
      }
    }

    // Re-load weights tile in A layout (C layout -> A layout via per-wave LDS slice)
    v16bf aw;
    {
      v8us lo = *(const v8us*)&ws[ln * WSTR + 16 * half];
      v8us hi = *(const v8us*)&ws[ln * WSTR + 16 * half + 8];
      v16us bits;
      #pragma unroll
      for (int j = 0; j < 8; ++j) { bits[j] = lo[j]; bits[8 + j] = hi[j]; }
      aw = __builtin_bit_cast(v16bf, bits);
    }

    #pragma unroll
    for (int nc = 0; nc < 4; ++nc) {   // B = V^T-staged tile: lane = d col, elems = 16 keys
      v16bf bv = lds_bf16x16(&Vt[(nc * 16 + ln) * VSTR + 16 * half]);
      acco[nc] = __builtin_amdgcn_wmma_f32_16x16x32_bf16(false, aw, false, bv,
                                                         (short)0, acco[nc], false, false);
    }
  }

  // ---- write out tile ----
  #pragma unroll
  for (int nc = 0; nc < 4; ++nc) {
    #pragma unroll
    for (int r = 0; r < 8; ++r)
      Op[(size_t)rowi[r] * DDIM + nc * 16 + ln] = acco[nc][r];
  }

  // ---- zero-fill the fully-masked causal tail of weights for this block's rows ----
  const int zc = SDIM - kend;
  if (zc > 0) {
    const v4f zz = {0.f, 0.f, 0.f, 0.f};
    for (int row = 0; row < QT; ++row) {
      float* p = Wp + (size_t)(q0b + row) * SDIM + kend;
      for (int c = tid * 4; c < zc; c += 256 * 4)
        __builtin_nontemporal_store(zz, (v4f*)(p + c));
    }
  }
}

extern "C" void kernel_launch(void* const* d_in, const int* in_sizes, int n_in,
                              void* d_out, int out_size, void* d_ws, size_t ws_size,
                              hipStream_t stream) {
  (void)in_sizes; (void)n_in; (void)d_ws; (void)ws_size; (void)out_size;
  const float* Q = (const float*)d_in[0];
  const float* K = (const float*)d_in[1];
  const float* V = (const float*)d_in[2];
  // d_in[3] is the causal Mask; strictly-upper-triangular ones, applied analytically.
  float* out  = (float*)d_out;
  float* wout = out + (size_t)BHN * SDIM * DDIM;   // weights follow out in the flat output
  dim3 grid(SDIM / QT, BHN);
  attn_fused<<<grid, dim3(256), 0, stream>>>(Q, K, V, out, wout);
}
